// MultiheadAttention_90855738179833
// MI455X (gfx1250) — matrix operand order repaired and checked by reading, not compiled, not run on hardware
//
#include <hip/hip_runtime.h>
#include <hip/hip_bf16.h>

// ---------------------------------------------------------------------------
// MHA for MI455X (gfx1250, wave32). All GEMMs on v_wmma_f32_16x16x32_f16.
// Flash attention with TDM (tensor_load_to_lds) double-buffered K/V staging.
// ---------------------------------------------------------------------------

typedef _Float16 half_t;
typedef __attribute__((ext_vector_type(16))) _Float16 v16h;
typedef __attribute__((ext_vector_type(8)))  _Float16 v8h;
typedef __attribute__((ext_vector_type(8)))  float    v8f;
typedef __attribute__((ext_vector_type(4)))  float    v4f;
typedef __attribute__((ext_vector_type(4)))  unsigned int u32x4;
typedef __attribute__((ext_vector_type(8)))  int      i32x8;
typedef __attribute__((ext_vector_type(4)))  int      i32x4;

#define BB 4
#define SS 2048
#define DD 1024
#define HH 16
#define DKV 64

__device__ __forceinline__ v8f wmma_f16(v16h a, v16h b, v8f c) {
  return __builtin_amdgcn_wmma_f32_16x16x32_f16(
      false, a, false, b, (short)0, c, false, false);
}

__device__ __forceinline__ v16h join8(v8h lo, v8h hi) {
  v16h r;
#pragma unroll
  for (int i = 0; i < 8; ++i) { r[i] = lo[i]; r[i + 8] = hi[i]; }
  return r;
}

__device__ __forceinline__ v8h cvt8(v4f a, v4f b) {
  v8h r;
#pragma unroll
  for (int i = 0; i < 4; ++i) { r[i] = (half_t)a[i]; r[i + 4] = (half_t)b[i]; }
  return r;
}

// A fragment (16x32 f16) from fp32 row: per-lane runs at half*8 and 16+half*8
__device__ __forceinline__ v16h load_a_f32(const float* __restrict__ row, int k, int hl) {
  const float* p0 = row + k + hl * 8;
  const float* p1 = row + k + 16 + hl * 8;
  v4f a0 = *(const v4f*)p0;       v4f a1 = *(const v4f*)(p0 + 4);
  v4f b0 = *(const v4f*)p1;       v4f b1 = *(const v4f*)(p1 + 4);
  return join8(cvt8(a0, a1), cvt8(b0, b1));
}

__device__ __forceinline__ v16h load_a_f16(const half_t* __restrict__ row, int k, int hl) {
  return join8(*(const v8h*)(row + k + hl * 8),
               *(const v8h*)(row + k + 16 + hl * 8));
}

// B fragment (32x16 f16) from [N][K] f16 row: per-lane runs at half*8 and 16+half*8
__device__ __forceinline__ v16h load_b_f16(const half_t* row, int k, int hl) {
  const half_t* p = row + k + hl * 8;
  return join8(*(const v8h*)p, *(const v8h*)(p + 16));
}

// ---------------------------------------------------------------------------
// TDM: 2D f16 tile load Global->LDS. D# built per CDNA5 ISA ch.8.
// Padding gives a bank-conflict-free LDS row pitch.
// 6-arg builtin form: (g0 u32x4, g1 i32x8, g2 i32x4, g3 i32x4, gx i32x8, cpol)
// ---------------------------------------------------------------------------
__device__ __forceinline__ void tdm_load_2d_f16(
    unsigned lds_off, const half_t* gptr,
    unsigned tdim0, unsigned tdim1, unsigned long long tstride0,
    unsigned tile0, unsigned tile1,
    unsigned pad_interval, unsigned pad_amount, unsigned pad_en) {
  unsigned long long ga = (unsigned long long)(const void*)gptr;
  u32x4 g0;
  g0[0] = 1u;                                   // count=1 (valid), user mode
  g0[1] = lds_off;                              // lds_addr (bytes)
  g0[2] = (unsigned)ga;                         // global_addr[31:0]
  g0[3] = (unsigned)((ga >> 32) & 0x01ffffffu) | (2u << 30);  // addr[56:32], type=2
  i32x8 g1;
  g1[0] = (int)((1u << 16) | (pad_en << 20) | (pad_interval << 22) | (pad_amount << 25));
  g1[1] = (int)((tdim0 & 0xffffu) << 16);                         // tensor_dim0 lo
  g1[2] = (int)(((tdim0 >> 16) & 0xffffu) | ((tdim1 & 0xffffu) << 16));
  g1[3] = (int)(((tdim1 >> 16) & 0xffffu) | ((tile0 & 0xffffu) << 16));
  g1[4] = (int)(tile1 & 0xffffu);                                 // tile_dim1
  g1[5] = (int)(unsigned)(tstride0 & 0xffffffffu);                // dim0 stride lo
  g1[6] = (int)(unsigned)((tstride0 >> 32) & 0xffffu);            // dim0 stride hi
  g1[7] = 0;
  i32x4 z4 = {0, 0, 0, 0};
  i32x8 z8 = {0, 0, 0, 0, 0, 0, 0, 0};
  __builtin_amdgcn_tensor_load_to_lds(g0, g1, z4, z4, z8, 0);
}

// ---------------------------------------------------------------------------
// Kernel 0: fp32 -> f16 weight conversion (one-time, tiny).
// ---------------------------------------------------------------------------
__global__ __launch_bounds__(256) void cvt_kernel(const float* __restrict__ src,
                                                  half_t* __restrict__ dst, int n) {
  int i = (blockIdx.x * 256 + threadIdx.x) * 8;
  if (i < n) {
    v4f a = *(const v4f*)(src + i);
    v4f b = *(const v4f*)(src + i + 4);
    *(v8h*)(dst + i) = cvt8(a, b);
  }
}

// ---------------------------------------------------------------------------
// Kernel 1: projection GEMM C[8192x1024] = A_f32[8192x1024] * W_f16^T.
// 4 waves/block; wave = 16(M) x 64(N) tile; N-tile == one head.
// transposed==0 -> f16 [b,h,s,64]; transposed==1 -> f16 [b,h,64,s].
// ---------------------------------------------------------------------------
__global__ __launch_bounds__(128) void proj_kernel(
    const float* __restrict__ src, const half_t* __restrict__ wf,
    const float* __restrict__ bias, half_t* __restrict__ dst_qk,
    half_t* __restrict__ dst_vt, int transposed) {
  const int lane = threadIdx.x & 31;
  const int wv = threadIdx.x >> 5;
  const int l16 = lane & 15;
  const int hl = lane >> 4;

  const int mrow0 = blockIdx.x * 64 + wv * 16;
  const int h = blockIdx.y;
  const int b = mrow0 >> 11;
  const int s0 = mrow0 & 2047;

  const float* arow = src + (size_t)(mrow0 + l16) * DD;

  v8f acc[4] = {v8f{}, v8f{}, v8f{}, v8f{}};
  for (int k = 0; k < DD; k += 32) {
    v16h af = load_a_f32(arow, k, hl);
#pragma unroll
    for (int nt = 0; nt < 4; ++nt) {
      const half_t* wrow = wf + (size_t)(h * 64 + nt * 16 + l16) * DD;
      acc[nt] = wmma_f16(af, load_b_f16(wrow, k, hl), acc[nt]);
    }
  }

#pragma unroll
  for (int nt = 0; nt < 4; ++nt) {
    const int dk = nt * 16 + l16;
    const float bv = bias[h * 64 + dk];
    if (!transposed) {
      size_t base = (((size_t)(b * HH + h)) * SS + s0) * 64 + dk;
#pragma unroll
      for (int r = 0; r < 8; ++r)
        dst_qk[base + (size_t)(r + 8 * hl) * 64] = (half_t)(acc[nt][r] + bv);
    } else {
      v8h pk;
#pragma unroll
      for (int r = 0; r < 8; ++r) pk[r] = (half_t)(acc[nt][r] + bv);
      size_t off = ((size_t)(b * HH + h) * 64 + dk) * SS + s0 + 8 * hl;
      *(v8h*)(dst_vt + off) = pk;
    }
  }
}

// ---------------------------------------------------------------------------
// Kernel 2: flash attention. Wave = 16 query rows. Per 32-key block:
// TDM stages K(32x64) and V^T(64x32) into padded LDS (double-buffered),
// S = Q*K^T, online softmax, P -> A-fragment via LDS, O += P*V.
// ---------------------------------------------------------------------------
__global__ __launch_bounds__(128) void flash_kernel(
    const half_t* __restrict__ qh, const half_t* __restrict__ kh,
    const half_t* __restrict__ vt, half_t* __restrict__ oc) {
  // padded pitches: 72 halves (144B) / 40 halves (80B) -> conflict-free b128
  __shared__ __align__(16) half_t kbuf[2][32][72];
  __shared__ __align__(16) half_t vbuf[2][64][40];
  __shared__ __align__(16) half_t plds[4][16][40];

  const int lane = threadIdx.x & 31;
  const int wv = threadIdx.x >> 5;
  const int l16 = lane & 15;
  const int hl = lane >> 4;

  const int b = blockIdx.z;
  const int h = blockIdx.y;
  const int s0 = blockIdx.x * 64 + wv * 16;

  const half_t* qrow = qh + ((size_t)(b * HH + h) * SS + s0 + l16) * 64;
  const half_t* kbase = kh + (size_t)(b * HH + h) * SS * 64;
  const half_t* vbase = vt + (size_t)(b * HH + h) * 64 * SS;

  const v16h qa0 = load_a_f16(qrow, 0, hl);
  const v16h qa1 = load_a_f16(qrow, 32, hl);

  float mrun[8], lrun[8];
#pragma unroll
  for (int r = 0; r < 8; ++r) { mrun[r] = -__builtin_inff(); lrun[r] = 0.0f; }
  v8f o[4] = {v8f{}, v8f{}, v8f{}, v8f{}};

  // prologue: stage t0=0 into buffer 0 (wave 0 only; TENSORcnt tracks it)
  if (wv == 0) {
    tdm_load_2d_f16((unsigned)(unsigned long long)(void*)&kbuf[0][0][0],
                    kbase, 64, SS, 64, 64, 32, /*pad 32dw by 4dw*/ 4, 3, 1);
    tdm_load_2d_f16((unsigned)(unsigned long long)(void*)&vbuf[0][0][0],
                    vbase, SS, 64, SS, 32, 64, /*pad 16dw by 4dw*/ 3, 3, 1);
  }

  for (int t0 = 0; t0 < SS; t0 += 32) {
    const int cur = (t0 >> 5) & 1;
    if (wv == 0) {
      if (t0 + 32 < SS) {   // prefetch next block into the other buffer
        tdm_load_2d_f16((unsigned)(unsigned long long)(void*)&kbuf[cur ^ 1][0][0],
                        kbase + (size_t)(t0 + 32) * 64, 64, SS, 64, 64, 32, 4, 3, 1);
        tdm_load_2d_f16((unsigned)(unsigned long long)(void*)&vbuf[cur ^ 1][0][0],
                        vbase + (t0 + 32), SS, 64, SS, 32, 64, 3, 3, 1);
        __builtin_amdgcn_s_wait_tensorcnt(2);  // current buffer's 2 loads done
      } else {
        __builtin_amdgcn_s_wait_tensorcnt(0);
      }
    }
    __syncthreads();  // current K/V tiles visible to all waves

    // ---- scores: 16x32 = Q(16x64) x K^T, K from LDS -------------------
    v8f sc[2];
#pragma unroll
    for (int nt = 0; nt < 2; ++nt) {
      const half_t* kr = &kbuf[cur][nt * 16 + l16][0];
      v8f z = v8f{};
      z = wmma_f16(qa0, load_b_f16(kr, 0, hl), z);
      z = wmma_f16(qa1, load_b_f16(kr, 32, hl), z);
      sc[nt] = z;
    }
#pragma unroll
    for (int nt = 0; nt < 2; ++nt)
#pragma unroll
      for (int r = 0; r < 8; ++r) sc[nt][r] *= 0.125f;  // 1/sqrt(64)

    // ---- online softmax ------------------------------------------------
    float mt[8];
#pragma unroll
    for (int r = 0; r < 8; ++r) mt[r] = fmaxf(sc[0][r], sc[1][r]);
#pragma unroll
    for (int msk = 1; msk < 16; msk <<= 1)
#pragma unroll
      for (int r = 0; r < 8; ++r) mt[r] = fmaxf(mt[r], __shfl_xor(mt[r], msk, 32));

    float alpha[8], rs[8];
#pragma unroll
    for (int r = 0; r < 8; ++r) {
      float mnew = fmaxf(mrun[r], mt[r]);
      alpha[r] = __expf(mrun[r] - mnew);
      mrun[r] = mnew;
      float p0 = __expf(sc[0][r] - mnew);
      float p1 = __expf(sc[1][r] - mnew);
      sc[0][r] = p0; sc[1][r] = p1;
      rs[r] = p0 + p1;
    }
#pragma unroll
    for (int msk = 1; msk < 16; msk <<= 1)
#pragma unroll
      for (int r = 0; r < 8; ++r) rs[r] += __shfl_xor(rs[r], msk, 32);
#pragma unroll
    for (int r = 0; r < 8; ++r) lrun[r] = lrun[r] * alpha[r] + rs[r];
#pragma unroll
    for (int nt = 0; nt < 4; ++nt)
#pragma unroll
      for (int r = 0; r < 8; ++r) o[nt][r] *= alpha[r];

    // ---- P: C-layout -> A-fragment via LDS -----------------------------
#pragma unroll
    for (int nt = 0; nt < 2; ++nt)
#pragma unroll
      for (int r = 0; r < 8; ++r)
        plds[wv][r + 8 * hl][nt * 16 + l16] = (half_t)sc[nt][r];
    __syncthreads();
    v16h pa = join8(*(const v8h*)&plds[wv][l16][hl * 8],
                    *(const v8h*)&plds[wv][l16][hl * 8 + 16]);

    // ---- O += P(16x32) x V(32x64), V^T tile from LDS -------------------
#pragma unroll
    for (int nt = 0; nt < 4; ++nt) {
      const half_t* vr = &vbuf[cur][nt * 16 + l16][hl * 8];
      v16h vb = join8(*(const v8h*)vr, *(const v8h*)(vr + 16));
      o[nt] = wmma_f16(pa, vb, o[nt]);
    }
    __syncthreads();  // all reads of buf[cur] + plds done before reuse
  }

  // ---- epilogue: normalize, store oc[b][s][h*64+dv] ----------------------
#pragma unroll
  for (int r = 0; r < 8; ++r) lrun[r] = 1.0f / lrun[r];
#pragma unroll
  for (int nt = 0; nt < 4; ++nt) {
    const int col = h * 64 + nt * 16 + l16;
#pragma unroll
    for (int r = 0; r < 8; ++r) {
      size_t row = (size_t)b * SS + s0 + r + 8 * hl;
      oc[row * (HH * DKV) + col] = (half_t)(o[nt][r] * lrun[r]);
    }
  }
}

// ---------------------------------------------------------------------------
// Kernel 3: output projection. out[8192x1024] = oc_f16 x Wo_f16^T + b (f32).
// ---------------------------------------------------------------------------
__global__ __launch_bounds__(128) void oproj_kernel(
    const half_t* __restrict__ oc, const half_t* __restrict__ wof,
    const float* __restrict__ wob, float* __restrict__ out) {
  const int lane = threadIdx.x & 31;
  const int wv = threadIdx.x >> 5;
  const int l16 = lane & 15;
  const int hl = lane >> 4;

  const int mrow0 = blockIdx.x * 64 + wv * 16;
  const int n0 = blockIdx.y * 64;

  const half_t* arow = oc + (size_t)(mrow0 + l16) * (HH * DKV);

  v8f acc[4] = {v8f{}, v8f{}, v8f{}, v8f{}};
  for (int k = 0; k < HH * DKV; k += 32) {
    v16h af = load_a_f16(arow, k, hl);
#pragma unroll
    for (int nt = 0; nt < 4; ++nt) {
      const half_t* wrow = wof + (size_t)(n0 + nt * 16 + l16) * (HH * DKV);
      acc[nt] = wmma_f16(af, load_b_f16(wrow, k, hl), acc[nt]);
    }
  }

#pragma unroll
  for (int nt = 0; nt < 4; ++nt) {
    const int n = n0 + nt * 16 + l16;
    const float bv = wob[n];
#pragma unroll
    for (int r = 0; r < 8; ++r)
      out[(size_t)(mrow0 + r + 8 * hl) * DD + n] = acc[nt][r] + bv;
  }
}

// ---------------------------------------------------------------------------
// Workspace (f16): qh | kh | vt | oc  (4 x 16MB) then wq|wk|wv|wo f16 (4 x 2MB).
// ---------------------------------------------------------------------------
extern "C" void kernel_launch(void* const* d_in, const int* in_sizes, int n_in,
                              void* d_out, int out_size, void* d_ws, size_t ws_size,
                              hipStream_t stream) {
  (void)in_sizes; (void)n_in; (void)out_size; (void)ws_size;
  const float* q    = (const float*)d_in[0];
  const float* k    = (const float*)d_in[1];
  const float* v    = (const float*)d_in[2];
  const float* wq_w = (const float*)d_in[3];
  const float* wq_b = (const float*)d_in[4];
  const float* wk_w = (const float*)d_in[5];
  const float* wk_b = (const float*)d_in[6];
  const float* wv_w = (const float*)d_in[7];
  const float* wv_b = (const float*)d_in[8];
  const float* wo_w = (const float*)d_in[9];
  const float* wo_b = (const float*)d_in[10];

  const size_t SEG = (size_t)BB * HH * SS * DKV;  // 8388608
  const int WN = DD * DD;                         // 1048576
  half_t* qh  = (half_t*)d_ws;
  half_t* kh  = qh + SEG;
  half_t* vt  = kh + SEG;
  half_t* oc  = vt + SEG;
  half_t* wqf = oc + SEG;
  half_t* wkf = wqf + WN;
  half_t* wvf = wkf + WN;
  half_t* wof = wvf + WN;

  dim3 blk(128);
  // one-time weight conversion fp32 -> f16
  cvt_kernel<<<dim3(WN / (256 * 8)), dim3(256), 0, stream>>>(wq_w, wqf, WN);
  cvt_kernel<<<dim3(WN / (256 * 8)), dim3(256), 0, stream>>>(wk_w, wkf, WN);
  cvt_kernel<<<dim3(WN / (256 * 8)), dim3(256), 0, stream>>>(wv_w, wvf, WN);
  cvt_kernel<<<dim3(WN / (256 * 8)), dim3(256), 0, stream>>>(wo_w, wof, WN);
  // QKV projections
  proj_kernel<<<dim3(128, 16), blk, 0, stream>>>(q, wqf, wq_b, qh, nullptr, 0);
  proj_kernel<<<dim3(128, 16), blk, 0, stream>>>(k, wkf, wk_b, kh, nullptr, 0);
  proj_kernel<<<dim3(128, 16), blk, 0, stream>>>(v, wvf, wv_b, nullptr, vt, 1);
  // Flash attention: grid = (S/64, H, B)
  flash_kernel<<<dim3(32, 16, 4), blk, 0, stream>>>(qh, kh, vt, oc);
  // Output projection
  oproj_kernel<<<dim3(128, 16), blk, 0, stream>>>(oc, wof, wo_b, (float*)d_out);
}
